// PredictModelWithCLasses_44418551775739
// MI455X (gfx1250) — compile-verified, matrix operand order checked by reading, not run on hardware
//
#include <hip/hip_runtime.h>
#include <stdint.h>

// ---------------------------------------------------------------------------
// Problem constants
// ---------------------------------------------------------------------------
#define BATCH   4096
#define NPROT   64
#define SPOT    512
#define NCLS    32
#define N1      544          // trans output dim (= INPUT_DIM)
#define KT1F    33           // fused B matrix K-tiles (1056/32): [0,16) pe | [16,33) spot
#define NT1     34           // 544/16
#define KT1     16           // main GEMM1 K-tiles (pe part only, 512/32)
#define KTS     17           // spot GEMM K-tiles (544/32)
#define KT2     17           // 544/32
#define NT2     9            // 144/16 (136 padded to 144)
#define H1DIM   136
#define H1PAD   160          // K for GEMM3 padded to 32-multiple
#define KT3     5            // 160/32
#define NT3     17           // 272/16
#define H2DIM   272
#define TILE_M  64           // main kernel batch rows per workgroup
#define TILE_MS 32           // spot kernel batch rows per workgroup

typedef __attribute__((ext_vector_type(16))) __bf16 v16bf;
typedef __attribute__((ext_vector_type(8)))  float  v8f;

// All loads of a kt-iteration are issued before this; WMMAs after.
// sched_barrier(0): no instruction may be moved across by the scheduler.
#define LOAD_COMPUTE_FENCE() __builtin_amdgcn_sched_barrier(0)

// ---------------------------------------------------------------------------
// bf16 helpers (round-to-nearest-even)
// ---------------------------------------------------------------------------
__device__ __forceinline__ unsigned short f2bf(float f) {
    union { float f; uint32_t u; } x; x.f = f;
    uint32_t r = x.u + 0x7FFFu + ((x.u >> 16) & 1u);
    return (unsigned short)(r >> 16);
}
__device__ __forceinline__ float bf2f(unsigned short h) {
    union { uint32_t u; float f; } x; x.u = ((uint32_t)h) << 16;
    return x.f;
}
__device__ __forceinline__ uint32_t pack2bf(float f0, float f1) {
    return (uint32_t)f2bf(f0) | ((uint32_t)f2bf(f1) << 16);
}

// Gather one 16x32 bf16 A fragment from row-major bf16 LDS.
// ISA 7.12.2 16-bit A layout: dword v holds the K-pair at
//   koff = (v<4?0:16) + (lane/16)*8 + 2*(v%4)
__device__ __forceinline__ v16bf load_a_frag(const uint32_t* lds32,
                                             int baseDW, int ktOff, int half4) {
    union { v16bf v; uint32_t u[8]; } a;
#pragma unroll
    for (int v = 0; v < 8; ++v) {
        int hv = ((v < 4) ? 0 : 8) + half4 + (v & 3);
        a.u[v] = lds32[baseDW + ktOff + hv];
    }
    return a.v;
}

// Pre-swizzled B fragment: fragIdx*1KB + lane*32B, 32-bit offset so the
// backend selects SADDR + voffset global_load_b128 pairs.
__device__ __forceinline__ v16bf load_b_frag(const unsigned short* base,
                                             unsigned fragIdx, unsigned lane) {
    unsigned off = fragIdx * 1024u + lane * 32u;
    return *(const v16bf*)((const char*)base + off);
}

#define WMMA_BF16(A, B, C) \
    __builtin_amdgcn_wmma_f32_16x16x32_bf16(false, (A), false, (B), (short)0, (C), false, false)

// ---------------------------------------------------------------------------
// Prep: fused B matrix [1056 x 544] -> bf16, pre-swizzled 32x16 fragments.
// rows 0..511   : combined_W = proj_W @ trans_W[0:512]   (used by main GEMM1)
// rows 512..1055: trans_W[512:1056]                      (used by spot GEMM)
// ---------------------------------------------------------------------------
__global__ __launch_bounds__(256) void build_wsB(const float* __restrict__ projW,
                                                 const float* __restrict__ transW,
                                                 unsigned short* __restrict__ wsB) {
    int tid = blockIdx.x * 256 + threadIdx.x;
    if (tid >= KT1F * NT1 * 512) return;
    int ktnt = tid >> 9, r = tid & 511;
    int lane = r >> 4, e = r & 15;
    int kt = ktnt / NT1, nt = ktnt % NT1;
    int half = lane >> 4, col = lane & 15;
    int vp = e >> 1, lo = e & 1;
    int k = kt * 32 + ((vp < 4) ? 0 : 16) + half * 8 + 2 * (vp & 3) + lo;
    int n = nt * 16 + col;
    float val;
    if (k < SPOT) {
        float acc = 0.f;
        for (int e2 = 0; e2 < SPOT; ++e2)
            acc += projW[k * SPOT + e2] * transW[e2 * N1 + n];
        val = acc;
    } else {
        val = transW[k * N1 + n];
    }
    wsB[tid] = f2bf(val);
}

// cvec[n] = trans_b[n] + proj_b . trans_W[0:512, n]
__global__ __launch_bounds__(256) void build_cvec(const float* __restrict__ projB,
                                                  const float* __restrict__ transW,
                                                  const float* __restrict__ transB,
                                                  float* __restrict__ cvec) {
    int n = blockIdx.x * 256 + threadIdx.x;
    if (n >= N1) return;
    float acc = transB[n];
    for (int e = 0; e < SPOT; ++e) acc += projB[e] * transW[e * N1 + n];
    cvec[n] = acc;
}

// W1: [64][544][136] -> bf16 fragments [p][17 kt][9 nt], N padded to 144 w/ 0
__global__ __launch_bounds__(256) void build_wsW1(const float* __restrict__ W1,
                                                  unsigned short* __restrict__ ws) {
    int tid = blockIdx.x * 256 + threadIdx.x;
    if (tid >= NPROT * KT2 * NT2 * 512) return;
    int per = KT2 * NT2 * 512;
    int p = tid / per, rem = tid % per;
    int ktnt = rem >> 9, r = rem & 511;
    int lane = r >> 4, e = r & 15;
    int kt = ktnt / NT2, nt = ktnt % NT2;
    int half = lane >> 4, col = lane & 15;
    int vp = e >> 1, lo = e & 1;
    int k = kt * 32 + ((vp < 4) ? 0 : 16) + half * 8 + 2 * (vp & 3) + lo;  // < 544
    int n = nt * 16 + col;                                                 // < 144
    float val = (n < H1DIM) ? W1[((size_t)p * N1 + k) * H1DIM + n] : 0.f;
    ws[tid] = f2bf(val);
}

// W2: [64][136][272] -> bf16 fragments [p][5 kt][17 nt], K padded to 160 w/ 0
__global__ __launch_bounds__(256) void build_wsW2(const float* __restrict__ W2,
                                                  unsigned short* __restrict__ ws) {
    int tid = blockIdx.x * 256 + threadIdx.x;
    if (tid >= NPROT * KT3 * NT3 * 512) return;
    int per = KT3 * NT3 * 512;
    int p = tid / per, rem = tid % per;
    int ktnt = rem >> 9, r = rem & 511;
    int lane = r >> 4, e = r & 15;
    int kt = ktnt / NT3, nt = ktnt % NT3;
    int half = lane >> 4, col = lane & 15;
    int vp = e >> 1, lo = e & 1;
    int k = kt * 32 + ((vp < 4) ? 0 : 16) + half * 8 + 2 * (vp & 3) + lo;  // < 160
    int n = nt * 16 + col;                                                 // < 272
    float val = (k < H1DIM) ? W2[((size_t)p * H1DIM + k) * H2DIM + n] : 0.f;
    ws[tid] = f2bf(val);
}

// ---------------------------------------------------------------------------
// spot GEMM: spotc[b][n] = [cell|cond][b] @ trans_W[512:1056] + cvec[n]
// Shared across all 64 proteins -> computed once per batch row.
// One workgroup = 32 rows, 256 threads = 8 waves (wm 0..1) x (wn 0..3).
// Uses kt 16..32 of the fused wsB fragments.  N-work split into two passes
// (5 + 4 tiles) to keep live accumulators <= 40 VGPRs per wave.
// ---------------------------------------------------------------------------
__global__ __launch_bounds__(256) void spot_gemm(
    const float* __restrict__ cell, const float* __restrict__ cond,
    const unsigned short* __restrict__ wsB, const float* __restrict__ cvec,
    float* __restrict__ spotc) {

    extern __shared__ char smem[];
    unsigned short* ldsS = (unsigned short*)smem;          // [32][544] bf16
    const uint32_t* ldsS32 = (const uint32_t*)ldsS;

    const int tid = threadIdx.x;
    const int b0 = blockIdx.x * TILE_MS;

    // build bf16 spot tile with float4 loads (544 = 512 cell + 32 cond)
    for (int t = tid; t < TILE_MS * (N1 / 4); t += 256) {
        int m = t / (N1 / 4);
        int q = t % (N1 / 4);
        int k = q * 4;
        int b = b0 + m;
        float4 v;
        if (k < SPOT) v = *(const float4*)&cell[(size_t)b * SPOT + k];
        else          v = *(const float4*)&cond[(size_t)b * NCLS + (k - SPOT)];
        uint32_t* d = (uint32_t*)ldsS + m * (N1 / 2) + q * 2;
        d[0] = pack2bf(v.x, v.y);
        d[1] = pack2bf(v.z, v.w);
    }
    __syncthreads();

    const int wave = tid >> 5, lane = tid & 31;
    const int wm = wave >> 2, wn = wave & 3;
    const int half = lane >> 4, col = lane & 15;
    const int ma = 16 * wm + col;
    const int half4 = half * 4;

    // pass 0: n-tiles wn+4j, j<5 (all < 34)
    {
        v8f acc[5] = {};
#pragma unroll 1
        for (int kt = 0; kt < KTS; ++kt) {
            v16bf bfr[5];
#pragma unroll
            for (int j = 0; j < 5; ++j)
                bfr[j] = load_b_frag(wsB, (unsigned)((kt + 16) * NT1 + wn + 4 * j), lane);
            v16bf a = load_a_frag(ldsS32, ma * (N1 / 2), kt * 16, half4);
            LOAD_COMPUTE_FENCE();
#pragma unroll
            for (int j = 0; j < 5; ++j)
                acc[j] = WMMA_BF16(a, bfr[j], acc[j]);
        }
#pragma unroll
        for (int j = 0; j < 5; ++j) {
            int n = (wn + 4 * j) * 16 + col;
            float bias = cvec[n];
#pragma unroll
            for (int r = 0; r < 8; ++r) {
                int m = 16 * wm + r + 8 * half;
                spotc[(size_t)(b0 + m) * N1 + n] = acc[j][r] + bias;
            }
        }
    }
    // pass 1: n-tiles wn+4(j+5), j<4 (clamp only on last)
    {
        v8f acc[4] = {};
#pragma unroll 1
        for (int kt = 0; kt < KTS; ++kt) {
            v16bf bfr[4];
#pragma unroll
            for (int j = 0; j < 4; ++j) {
                int nt = wn + 4 * (j + 5);
                int ntc = (nt < NT1) ? nt : 0;
                bfr[j] = load_b_frag(wsB, (unsigned)((kt + 16) * NT1 + ntc), lane);
            }
            v16bf a = load_a_frag(ldsS32, ma * (N1 / 2), kt * 16, half4);
            LOAD_COMPUTE_FENCE();
#pragma unroll
            for (int j = 0; j < 4; ++j)
                acc[j] = WMMA_BF16(a, bfr[j], acc[j]);
        }
#pragma unroll
        for (int j = 0; j < 4; ++j) {
            int nt = wn + 4 * (j + 5);
            if (nt < NT1) {
                int n = nt * 16 + col;
                float bias = cvec[n];
#pragma unroll
                for (int r = 0; r < 8; ++r) {
                    int m = 16 * wm + r + 8 * half;
                    spotc[(size_t)(b0 + m) * N1 + n] = acc[j][r] + bias;
                }
            }
        }
    }
}

// ---------------------------------------------------------------------------
// Fused main kernel: one workgroup = 64 batch rows x 1 protein.
// 512 threads = 16 wave32 laid out (wm in 0..3) x (wn in 0..3).
// GEMM1 (K=512, pe @ combined_W, two N-passes) + spotc -> GEMM2 -> GEMM3 -> GEMV
// ---------------------------------------------------------------------------
__global__ __launch_bounds__(512) void fused_main(
    const float* __restrict__ pe,
    const unsigned short* __restrict__ wsB, const float* __restrict__ spotc,
    const unsigned short* __restrict__ wsW1, const unsigned short* __restrict__ wsW2,
    const float* __restrict__ b1, const float* __restrict__ b2,
    const float* __restrict__ W3, const float* __restrict__ b3,
    float* __restrict__ out) {

    extern __shared__ char smem[];
    unsigned short* ldsA16  = (unsigned short*)smem;                // [64][512]  65536 B
    unsigned short* ldsX16  = (unsigned short*)(smem + 65536);      // [64][544]  69632 B
    unsigned short* ldsH116 = (unsigned short*)(smem + 135168);     // [64][160]  20480 B
    unsigned short* ldsH216 = (unsigned short*)(smem + 155648);     // [64][272]  34816 B
    const uint32_t* ldsA32  = (const uint32_t*)ldsA16;
    const uint32_t* ldsX32  = (const uint32_t*)ldsX16;
    const uint32_t* ldsH132 = (const uint32_t*)ldsH116;

    const int tid = threadIdx.x;
    const int p   = blockIdx.y;            // 0..63
    const int b0  = blockIdx.x * TILE_M;   // 0..4032

    // ---- Stage 0: bf16(pe) tile [64][512] via float4 loads (HBM stream) ----
    for (int t = tid; t < TILE_M * (SPOT / 4); t += 512) {
        int m = t >> 7;                     // SPOT/4 = 128
        int q = t & 127;
        float4 v = *((const float4*)&pe[((size_t)(b0 + m) * NPROT + p) * SPOT] + q);
        uint32_t* d = (uint32_t*)ldsA16 + m * (SPOT / 2) + q * 2;
        d[0] = pack2bf(v.x, v.y);
        d[1] = pack2bf(v.z, v.w);
    }
    __syncthreads();

    const int wave = tid >> 5, lane = tid & 31;
    const int wm = wave >> 2, wn = wave & 3;
    const int half = lane >> 4, col = lane & 15;
    const int ma = 16 * wm + col;          // A-fragment row for this lane
    const int half4 = half * 4;

    // ---- GEMM1: x[64][544] = peTile @ combined_W + spotc -------------------
    // pass 0: n-tiles wn+4j, j<5 (all < 34, no clamp, no wasted work)
    {
        v8f acc[5] = {};
#pragma unroll 1
        for (int kt = 0; kt < KT1; ++kt) {
            v16bf bfr[5];
#pragma unroll
            for (int j = 0; j < 5; ++j)
                bfr[j] = load_b_frag(wsB, (unsigned)(kt * NT1 + wn + 4 * j), lane);
            v16bf a = load_a_frag(ldsA32, ma * (SPOT / 2), kt * 16, half4);
            LOAD_COMPUTE_FENCE();
#pragma unroll
            for (int j = 0; j < 5; ++j)
                acc[j] = WMMA_BF16(a, bfr[j], acc[j]);
        }
#pragma unroll
        for (int j = 0; j < 5; ++j) {
            int n = (wn + 4 * j) * 16 + col;
#pragma unroll
            for (int r = 0; r < 8; ++r) {
                int m = 16 * wm + r + 8 * half;     // C/D layout: lanes16-31 hold M=r+8
                float x = acc[j][r] + spotc[(size_t)(b0 + m) * N1 + n];
                ldsX16[m * N1 + n] = f2bf(x);
            }
        }
    }
    // pass 1: n-tiles wn+4(j+5), j<4 (clamp only on last tile of wn 2,3)
    {
        v8f acc[4] = {};
#pragma unroll 1
        for (int kt = 0; kt < KT1; ++kt) {
            v16bf bfr[4];
#pragma unroll
            for (int j = 0; j < 4; ++j) {
                int nt = wn + 4 * (j + 5);
                int ntc = (nt < NT1) ? nt : 0;
                bfr[j] = load_b_frag(wsB, (unsigned)(kt * NT1 + ntc), lane);
            }
            v16bf a = load_a_frag(ldsA32, ma * (SPOT / 2), kt * 16, half4);
            LOAD_COMPUTE_FENCE();
#pragma unroll
            for (int j = 0; j < 4; ++j)
                acc[j] = WMMA_BF16(a, bfr[j], acc[j]);
        }
#pragma unroll
        for (int j = 0; j < 4; ++j) {
            int nt = wn + 4 * (j + 5);
            if (nt < NT1) {
                int n = nt * 16 + col;
#pragma unroll
                for (int r = 0; r < 8; ++r) {
                    int m = 16 * wm + r + 8 * half;
                    float x = acc[j][r] + spotc[(size_t)(b0 + m) * N1 + n];
                    ldsX16[m * N1 + n] = f2bf(x);
                }
            }
        }
    }
    __syncthreads();

    // ---- GEMM2: h1[64][144] = leaky(x @ W1_p + b1) -------------------------
    {
        const unsigned short* W1f = wsW1 + (size_t)p * (KT2 * NT2 * 512);
        v8f acc[3] = {};
#pragma unroll 1
        for (int kt = 0; kt < KT2; ++kt) {
            v16bf bfr[3];
#pragma unroll
            for (int j = 0; j < 3; ++j) {
                int nt = wn + 4 * j;
                int ntc = (nt < NT2) ? nt : 0;
                bfr[j] = load_b_frag(W1f, (unsigned)(kt * NT2 + ntc), lane);
            }
            v16bf a = load_a_frag(ldsX32, ma * (N1 / 2), kt * 16, half4);
            LOAD_COMPUTE_FENCE();
#pragma unroll
            for (int j = 0; j < 3; ++j)
                acc[j] = WMMA_BF16(a, bfr[j], acc[j]);
        }
#pragma unroll
        for (int j = 0; j < 3; ++j) {
            int nt = wn + 4 * j;
            if (nt < NT2) {
                int n = nt * 16 + col;
                float bias = (n < H1DIM) ? b1[p * H1DIM + n] : 0.f;
#pragma unroll
                for (int r = 0; r < 8; ++r) {
                    int m = 16 * wm + r + 8 * half;
                    float v = acc[j][r] + bias;
                    v = (v > 0.f) ? v : 0.2f * v;
                    ldsH116[m * H1PAD + n] = f2bf(v);
                }
            }
        }
        // zero-fill padded K columns 144..159 for GEMM3
        for (int i = tid; i < TILE_M * 16; i += 512) {
            int m = i >> 4, c = 144 + (i & 15);
            ldsH116[m * H1PAD + c] = 0;
        }
    }
    __syncthreads();

    // ---- GEMM3: h2[64][272] = leaky(h1 @ W2_p + b2) ------------------------
    {
        const unsigned short* W2f = wsW2 + (size_t)p * (KT3 * NT3 * 512);
        v8f acc[5] = {};
#pragma unroll 1
        for (int kt = 0; kt < KT3; ++kt) {
            v16bf bfr[5];
#pragma unroll
            for (int j = 0; j < 5; ++j) {
                int nt = wn + 4 * j;
                int ntc = (nt < NT3) ? nt : 0;
                bfr[j] = load_b_frag(W2f, (unsigned)(kt * NT3 + ntc), lane);
            }
            v16bf a = load_a_frag(ldsH132, ma * (H1PAD / 2), kt * 16, half4);
            LOAD_COMPUTE_FENCE();
#pragma unroll
            for (int j = 0; j < 5; ++j)
                acc[j] = WMMA_BF16(a, bfr[j], acc[j]);
        }
#pragma unroll
        for (int j = 0; j < 5; ++j) {
            int nt = wn + 4 * j;
            if (nt < NT3) {
                int n = nt * 16 + col;
                float bias = b2[p * H2DIM + n];
#pragma unroll
                for (int r = 0; r < 8; ++r) {
                    int m = 16 * wm + r + 8 * half;
                    float v = acc[j][r] + bias;
                    v = (v > 0.f) ? v : 0.2f * v;
                    ldsH216[m * H2DIM + n] = f2bf(v);
                }
            }
        }
    }
    __syncthreads();

    // ---- GEMV: out[m] = h2[m] . W3_p + b3_p  (8 threads per row) ----------
    {
        int m = tid >> 3, g = tid & 7;     // 64 rows x 8 threads
        float part = 0.f;
        for (int k = g; k < H2DIM; k += 8)
            part += bf2f(ldsH216[m * H2DIM + k]) * W3[p * H2DIM + k];
        part += __shfl_xor(part, 1);
        part += __shfl_xor(part, 2);
        part += __shfl_xor(part, 4);
        if (g == 0) out[(size_t)(b0 + m) * NPROT + p] = part + b3[p];
    }
}

// ---------------------------------------------------------------------------
// Launch
// ---------------------------------------------------------------------------
extern "C" void kernel_launch(void* const* d_in, const int* in_sizes, int n_in,
                              void* d_out, int out_size, void* d_ws, size_t ws_size,
                              hipStream_t stream) {
    const float* cell   = (const float*)d_in[0];
    const float* cond   = (const float*)d_in[1];
    const float* pe     = (const float*)d_in[2];
    const float* projW  = (const float*)d_in[3];
    const float* projB  = (const float*)d_in[4];
    const float* transW = (const float*)d_in[5];
    const float* transB = (const float*)d_in[6];
    const float* W1     = (const float*)d_in[7];
    const float* b1     = (const float*)d_in[8];
    const float* W2     = (const float*)d_in[9];
    const float* b2     = (const float*)d_in[10];
    const float* W3     = (const float*)d_in[11];
    const float* b3     = (const float*)d_in[12];
    float* out = (float*)d_out;

    // workspace layout (all 32B aligned)
    char* ws = (char*)d_ws;
    unsigned short* wsB   = (unsigned short*)(ws);              //  1,148,928 B
    float*          cvec  = (float*)(ws + 1148928);             //      2,176 B
    unsigned short* wsW1  = (unsigned short*)(ws + 1151104);    // 10,027,008 B
    unsigned short* wsW2  = (unsigned short*)(ws + 11178112);   //  5,570,560 B
    float*          spotc = (float*)(ws + 16748672);            //  8,912,896 B -> 25,661,568 total

    {
        int n = KT1F * NT1 * 512;
        build_wsB<<<(n + 255) / 256, 256, 0, stream>>>(projW, transW, wsB);
    }
    build_cvec<<<3, 256, 0, stream>>>(projB, transW, transB, cvec);
    {
        int n = NPROT * KT2 * NT2 * 512;
        build_wsW1<<<(n + 255) / 256, 256, 0, stream>>>(W1, wsW1);
    }
    {
        int n = NPROT * KT3 * NT3 * 512;
        build_wsW2<<<(n + 255) / 256, 256, 0, stream>>>(W2, wsW2);
    }
    // spotc depends only on build_wsB + build_cvec (same stream -> ordered)
    spot_gemm<<<BATCH / TILE_MS, 256, TILE_MS * N1 * 2, stream>>>(cell, cond, wsB, cvec, spotc);

    dim3 grid(BATCH / TILE_M, NPROT);
    fused_main<<<grid, 512, 190464, stream>>>(pe, wsB, spotc, wsW1, wsW2,
                                              b1, b2, W3, b3, out);
}
